// tCross_27668179320888
// MI455X (gfx1250) — matrix operand order repaired and checked by reading, not compiled, not run on hardware
//
#include <hip/hip_runtime.h>
#include <math.h>

typedef __attribute__((ext_vector_type(16))) _Float16 v16h;
typedef __attribute__((ext_vector_type(8)))  _Float16 v8h;
typedef __attribute__((ext_vector_type(4)))  _Float16 v4h;
typedef __attribute__((ext_vector_type(8)))  float    v8f;

#define DD   128          // feature dim
#define RPB  128          // rows per block
#define THREADS 256       // 8 waves (wave32)
#define LWSTRIDE 144      // halves; 288B row stride: 32B-aligned, bank-friendly

// Branch-free GELU (tanh form). No EXEC divergence, ~10 VALU ops:
//   v_exp_f32 (native) + v_rcp_f32 + FMAs. Saturates correctly at +/-inf.
__device__ __forceinline__ float gelu_fast(float c) {
    float u = 0.7978845608028654f * fmaf(0.044715f * c, c * c, c);
    float e = __expf(2.0f * u);                               // v_exp_f32 path
    float t = fmaf(-2.0f, __builtin_amdgcn_rcpf(e + 1.0f), 1.0f); // tanh(u)
    return 0.5f * c * (1.0f + t);
}

__global__ __launch_bounds__(THREADS, 1)
void tCross_fused_kernel(const float* __restrict__ x,
                         const float* __restrict__ Wk, const float* __restrict__ bk,
                         const float* __restrict__ Wq, const float* __restrict__ bq,
                         const float* __restrict__ Wv, const float* __restrict__ bv,
                         float* __restrict__ out, int nrows)
{
    // LDS: one weight matrix (transposed, f16) + gelu(k|q|v) staging (f16)
    // Total 132 KB -> two workgroups fit per 320 KB WGP (occupancy 2).
    __shared__ __align__(32) _Float16 lw[DD * LWSTRIDE];     // 36 KB, lw[n*144+k] = W[k][n]
    __shared__ __align__(32) _Float16 lkqv[3][RPB * DD];     // 96 KB

    const int tid  = threadIdx.x;
    const int wave = tid >> 5;
    const int lane = tid & 31;
    const int li   = lane & 15;        // M (A) / N (B) index within tile
    const int hi   = lane >> 4;        // half-wave select
    const int row0 = blockIdx.x * RPB; // global row base of this block
    const int r0   = wave * 16;        // local row base of this wave

    // ---- Build A fragments once: 16 rows x 128 cols of x, f32 -> f16 ----
    // 16-bit A 16x32 layout: lane li holds row M=li; halves K = hi*16 + 0..15
    // (contiguous, matching B) per 32-wide K step kk.
    v16h afrag[4];
    {
        int ar = row0 + r0 + li;
        if (ar >= nrows) ar = nrows - 1;             // clamp: keep EXEC all-ones
        const float* xr = x + (size_t)ar * DD;
        #pragma unroll
        for (int kk = 0; kk < 4; ++kk) {
            const int kb = kk * 32 + hi * 16;
            #pragma unroll
            for (int i = 0; i < 8; ++i)
                afrag[kk][i] = (_Float16)xr[kb + i];
            #pragma unroll
            for (int i = 0; i < 8; ++i)
                afrag[kk][8 + i] = (_Float16)xr[kb + 8 + i];
        }
    }

    // ---- Stage 1: three projections + bias + GELU via WMMA f16 ----
    for (int m = 0; m < 3; ++m) {
        const float* W = (m == 0) ? Wk : (m == 1) ? Wq : Wv;
        const float* b = (m == 0) ? bk : (m == 1) ? bq : bv;

        // stage W^T as f16 into LDS (coalesced global read)
        for (int idx = tid; idx < DD * DD; idx += THREADS) {
            const int k = idx >> 7;
            const int n = idx & (DD - 1);
            lw[n * LWSTRIDE + k] = (_Float16)W[idx];
        }
        // per-lane bias for each of the 8 N-tiles (one b32 load each, L2-hot)
        float bias[8];
        #pragma unroll
        for (int nt = 0; nt < 8; ++nt) bias[nt] = b[nt * 16 + li];
        __syncthreads();

        #pragma unroll
        for (int nt = 0; nt < 8; ++nt) {
            const int n = nt * 16 + li;
            // B 32x16 layout: lane li holds col N=li; halves K = hi*16 + 0..15
            // (contiguous) per 32-wide K step -> straight 32B vector LDS loads.
            v8f acc = {};
            #pragma unroll
            for (int kk = 0; kk < 4; ++kk) {
                const v16h bfrag =
                    *(const v16h*)&lw[n * LWSTRIDE + kk * 32 + hi * 16];
                acc = __builtin_amdgcn_wmma_f32_16x16x32_f16(
                    false, afrag[kk], false, bfrag,
                    (short)0, acc, false, false);
            }
            // C/D layout: VGPR j holds (M = j + 8*hi, N = li)
            #pragma unroll
            for (int j = 0; j < 8; ++j) {
                float g = gelu_fast(acc[j] + bias[nt]);
                const int mrow = r0 + j + hi * 8;
                lkqv[m][mrow * DD + n] = (_Float16)g;
            }
        }
        __syncthreads();   // protect lw (WAR) and lkqv (RAW) across iterations
    }

    // ---- Stage 2: per-sample y = k^T q (8x8), res = v y (16x8), + residual ----
    // Column split: each thread of a row-pair owns 4 of the 8 g-columns, so it
    // only needs (and computes) its 4 columns of y -> no duplicated FMAs.
    {
        const int row  = tid >> 1;            // 0..127 local row
        const int part = tid & 1;
        const int g0   = part * 4;            // owned g-columns [g0, g0+4)
        const int grow = row0 + row;

        if (grow < nrows) {
            const _Float16* kp = &lkqv[0][row * DD];
            const _Float16* qp = &lkqv[1][row * DD];
            const _Float16* vp = &lkqv[2][row * DD];

            float y[8][4];                    // y[e][gg] for gg = g - g0
            #pragma unroll
            for (int e = 0; e < 8; ++e)
                #pragma unroll
                for (int gg = 0; gg < 4; ++gg) y[e][gg] = 0.0f;

            #pragma unroll
            for (int f = 0; f < 16; ++f) {
                const v8h kv = *(const v8h*)&kp[f * 8];        // ds_load_b128
                const v4h qv = *(const v4h*)&qp[f * 8 + g0];   // ds_load_b64
                float kf[8], qf[4];
                #pragma unroll
                for (int e = 0; e < 8; ++e) kf[e] = (float)kv[e];
                #pragma unroll
                for (int gg = 0; gg < 4; ++gg) qf[gg] = (float)qv[gg];
                #pragma unroll
                for (int e = 0; e < 8; ++e)
                    #pragma unroll
                    for (int gg = 0; gg < 4; ++gg)
                        y[e][gg] = fmaf(kf[e], qf[gg], y[e][gg]);
            }

            const float* xr   = x   + (size_t)grow * DD;
            float*       orow = out + (size_t)grow * DD;
            #pragma unroll
            for (int f = 0; f < 16; ++f) {
                const v8h vv = *(const v8h*)&vp[f * 8];        // ds_load_b128
                float vf[8];
                #pragma unroll
                for (int e = 0; e < 8; ++e) vf[e] = (float)vv[e];

                const float4 xv = *(const float4*)&xr[f * 8 + g0];  // b128
                float r[4];
                #pragma unroll
                for (int gg = 0; gg < 4; ++gg) {
                    float s = 0.0f;
                    #pragma unroll
                    for (int e = 0; e < 8; ++e) s = fmaf(vf[e], y[e][gg], s);
                    r[gg] = s;
                }
                float4 o = make_float4(r[0] + xv.x, r[1] + xv.y,
                                       r[2] + xv.z, r[3] + xv.w);
                *(float4*)&orow[f * 8 + g0] = o;               // b128 store
            }
        }
    }
}

extern "C" void kernel_launch(void* const* d_in, const int* in_sizes, int n_in,
                              void* d_out, int out_size, void* d_ws, size_t ws_size,
                              hipStream_t stream) {
    const float* x  = (const float*)d_in[0];
    const float* Wk = (const float*)d_in[1];
    const float* bk = (const float*)d_in[2];
    const float* Wq = (const float*)d_in[3];
    const float* bq = (const float*)d_in[4];
    const float* Wv = (const float*)d_in[5];
    const float* bv = (const float*)d_in[6];
    float* out = (float*)d_out;

    const int nrows = in_sizes[0] / DD;            // BATCH = 524288
    const int grid  = (nrows + RPB - 1) / RPB;     // 4096 blocks
    tCross_fused_kernel<<<grid, THREADS, 0, stream>>>(x, Wk, bk, Wq, bq, Wv, bv,
                                                      out, nrows);
}
